// HetGCNLayer_46952582480415
// MI455X (gfx1250) — matrix operand order repaired and checked by reading, not compiled, not run on hardware
//
#include <hip/hip_runtime.h>
#include <hip/hip_bf16.h>

// ---------------------------------------------------------------------------
// HetGCN layer for MI455X (gfx1250, wave32, WMMA).
// compat = rownorm(adj) @ x_b dominates: 8 x [2000x2000]@[2000x128], 8.2 GFLOP,
// 128 MB adj stream -> memory-bound at ~60 FLOP/B if matrix math >= 1.4 PFLOPS.
// f32 WMMA would be below that knee; bf16 16x16x32 WMMA (f32 accumulate) is
// comfortably above, so: stream adj f32, scale+convert to bf16 in registers,
// stage in LDS, v_wmma_f32_16x16x32_bf16. Two independent accumulators per
// wave (K split 64 = 2x32) break the WMMA RAW chain. Q/K/Wv chain fused into
// u = Wk^T Wv0, v = Wq^T Wv1 so the epilogue is one streaming pass.
// ---------------------------------------------------------------------------

typedef __attribute__((ext_vector_type(16))) __bf16 v16bf;
typedef __attribute__((ext_vector_type(8)))  float  v8f;

#define B_  8
#define NA  2000
#define NB  2000
#define DD  128
#define KT  64          // K per staging round = 2 WMMA chunks of 32
#define AP  72          // LDS row pitch in ushorts (144B: 16B-aligned, bank-clean)

static __device__ __forceinline__ unsigned short f32_bf16_rne(float f) {
  unsigned int u = __float_as_uint(f);
  u += 0x7FFFu + ((u >> 16) & 1u);       // round-to-nearest-even
  return (unsigned short)(u >> 16);
}
static __device__ __forceinline__ unsigned int pack_bf16(float lo, float hi) {
  return (unsigned int)f32_bf16_rne(lo) | ((unsigned int)f32_bf16_rne(hi) << 16);
}

// ---------------- Kernel 1: inverse row sums of adj (one wave per row) -----
__global__ __launch_bounds__(256) void rowsum_kernel(const float* __restrict__ adj,
                                                     float* __restrict__ inv) {
  int row  = blockIdx.x * (blockDim.x >> 5) + (threadIdx.x >> 5);
  int lane = threadIdx.x & 31;
  if (row >= B_ * NA) return;
  const float* p = adj + (size_t)row * NB;
  float s = 0.f;
  for (int c = lane; c < NB; c += 32) s += p[c];
  #pragma unroll
  for (int off = 16; off; off >>= 1) s += __shfl_xor(s, off, 32);
  if (lane == 0) inv[row] = (s == 0.f) ? 1.f : (1.f / s);
}

// ---------------- Kernel 2: u = Wk^T Wv[0,:D],  v = Wq^T Wv[0,D:] ----------
__global__ void uv_kernel(const float* __restrict__ Wq, const float* __restrict__ Wk,
                          const float* __restrict__ Wv, float* __restrict__ uv) {
  int d = threadIdx.x;
  if (d >= DD) return;
  float su = 0.f, sv = 0.f;
  for (int e = 0; e < DD; ++e) {
    su += Wk[e * DD + d] * Wv[e];
    sv += Wq[e * DD + d] * Wv[DD + e];
  }
  uv[d]      = su;
  uv[DD + d] = sv;
}

// ---------------- GEMM staging helper --------------------------------------
// Stages A (16 x 64, scaled, bf16) and B^T (128 x 64 as [d][k], bf16) tiles.
// GUARD=true only for the final partial round (k0 = 1984, 16 valid k).
template <bool GUARD>
static __device__ __forceinline__ void stage_tiles(
    const float* __restrict__ adjb, const float* __restrict__ xbb,
    float ascale, int k0, int tid,
    unsigned short* __restrict__ lds_a, unsigned short* __restrict__ lds_b) {
  // ---- A tile: 16 rows x 64 k, 4 contiguous f32 per thread ----
  {
    const int ar = tid >> 4;             // 0..15
    const int ac = (tid & 15) * 4;       // 0..60
    float x0, x1, x2, x3;
    if (!GUARD) {
      const float4 f = *(const float4*)(adjb + (size_t)ar * NB + k0 + ac);
      x0 = f.x; x1 = f.y; x2 = f.z; x3 = f.w;
    } else {
      const float* p = adjb + (size_t)ar * NB;
      x0 = (k0 + ac + 0 < NB) ? p[k0 + ac + 0] : 0.f;
      x1 = (k0 + ac + 1 < NB) ? p[k0 + ac + 1] : 0.f;
      x2 = (k0 + ac + 2 < NB) ? p[k0 + ac + 2] : 0.f;
      x3 = (k0 + ac + 3 < NB) ? p[k0 + ac + 3] : 0.f;
    }
    uint2 pk;
    pk.x = pack_bf16(x0 * ascale, x1 * ascale);
    pk.y = pack_bf16(x2 * ascale, x3 * ascale);
    *(uint2*)(lds_a + ar * AP + ac) = pk;   // one ds_store_b64
  }
  // ---- B tile: 64 k-rows x 128 d; thread owns k-pair {kr,kr+1} x 16 d ----
  // Stored transposed [d][k] so k-pairs pack into one b32 per d.
  {
    const int kr = (tid >> 3) * 2;       // 0,2,..,62
    const int c0 = (tid & 7) * 16;       // 0..112
    const float* r0 = xbb + (size_t)(k0 + kr) * DD + c0;
    const float* r1 = r0 + DD;
    #pragma unroll
    for (int j = 0; j < 4; ++j) {
      float4 a = {0.f, 0.f, 0.f, 0.f}, b = {0.f, 0.f, 0.f, 0.f};
      if (!GUARD) {
        a = *(const float4*)(r0 + j * 4);
        b = *(const float4*)(r1 + j * 4);
      } else {
        if (k0 + kr     < NB) a = *(const float4*)(r0 + j * 4);
        if (k0 + kr + 1 < NB) b = *(const float4*)(r1 + j * 4);
      }
      const int c = c0 + j * 4;
      *(unsigned int*)(lds_b + (c + 0) * AP + kr) = pack_bf16(a.x, b.x);
      *(unsigned int*)(lds_b + (c + 1) * AP + kr) = pack_bf16(a.y, b.y);
      *(unsigned int*)(lds_b + (c + 2) * AP + kr) = pack_bf16(a.z, b.z);
      *(unsigned int*)(lds_b + (c + 3) * AP + kr) = pack_bf16(a.w, b.w);
    }
  }
}

// ---------------- Kernel 3: compat = adj_norm @ x_b via bf16 WMMA ----------
// 256 threads = 8 waves; block owns rows [na0,na0+16) x all 128 cols.
// Wave w owns cols [16w,16w+16). Per round: K=64, 2 independent WMMAs.
__global__ __launch_bounds__(256) void gemm_kernel(const float* __restrict__ adj,
                                                   const float* __restrict__ xb,
                                                   const float* __restrict__ inv,
                                                   float* __restrict__ compat) {
  __shared__ unsigned short lds_a[16 * AP];    // 2.3 KB
  __shared__ unsigned short lds_b[DD * AP];    // 18.4 KB, transposed [d][k]

  const int b    = blockIdx.y;
  const int na0  = blockIdx.x * 16;
  const int tid  = threadIdx.x;
  const int lane = tid & 31;
  const int n0   = (tid >> 5) * 16;     // wave's D-column block
  const int mrow = lane & 15;
  const int half = lane >> 4;

  const float* adjb = adj + ((size_t)b * NA + na0) * NB;
  const float* xbb  = xb  +  (size_t)b * NB * DD;
  const float ascale = inv[(size_t)b * NA + na0 + (tid >> 4)];

  v8f acc0 = {}, acc1 = {};

  const uint4* arow = (const uint4*)(lds_a + mrow * AP);          // 144B rows
  const uint4* brow = (const uint4*)(lds_b + (n0 + mrow) * AP);

  int k0 = 0;
  #pragma unroll 1
  for (int r = 0; r < NB / KT; ++r, k0 += KT) {   // 31 full rounds
    stage_tiles<false>(adjb, xbb, ascale, k0, tid, lds_a, lds_b);
    __syncthreads();
    union { uint4 q[2]; v16bf v; } a0, a1, b0, b1;
    // A 16x32 frag, chunk c: v0..3 = K[32c+8h..+7], v4..7 = K[32c+16+8h..+7]
    a0.q[0] = arow[half];         a0.q[1] = arow[2 + half];
    a1.q[0] = arow[4 + half];     a1.q[1] = arow[6 + half];
    // B 32x16 frag, chunk c: v0..3 = K[32c+8h..+7], v4..7 = K[32c+16+8h..+7]
    b0.q[0] = brow[half];         b0.q[1] = brow[2 + half];
    b1.q[0] = brow[4 + half];     b1.q[1] = brow[6 + half];
    acc0 = __builtin_amdgcn_wmma_f32_16x16x32_bf16(false, a0.v, false, b0.v,
                                                   (short)0, acc0, false, false);
    acc1 = __builtin_amdgcn_wmma_f32_16x16x32_bf16(false, a1.v, false, b1.v,
                                                   (short)0, acc1, false, false);
    __syncthreads();
  }
  // tail round: k0 = 1984, 16 valid k (chunk1 stages as zeros)
  {
    stage_tiles<true>(adjb, xbb, ascale, k0, tid, lds_a, lds_b);
    __syncthreads();
    union { uint4 q[2]; v16bf v; } a0, b0;
    a0.q[0] = arow[half];     a0.q[1] = arow[2 + half];
    b0.q[0] = brow[half];     b0.q[1] = brow[2 + half];
    acc0 = __builtin_amdgcn_wmma_f32_16x16x32_bf16(false, a0.v, false, b0.v,
                                                   (short)0, acc0, false, false);
  }

  // C/D layout: VGPR r -> row (r + 8*half), col = lane&15 within block
  float* outp = compat + ((size_t)b * NA + na0) * DD + n0 + mrow;
  #pragma unroll
  for (int r = 0; r < 8; ++r)
    outp[(size_t)(r + 8 * half) * DD] = acc0[r] + acc1[r];
}

// ---------------- Kernel 4: fused ELU/softmax/blend epilogue ---------------
__global__ __launch_bounds__(256) void epilogue_kernel(const float* __restrict__ xa,
                                                       const float* __restrict__ compat,
                                                       const float* __restrict__ uv,
                                                       float* __restrict__ out) {
  int row  = blockIdx.x * (blockDim.x >> 5) + (threadIdx.x >> 5);
  int lane = threadIdx.x & 31;
  if (row >= B_ * NA) return;

  const float4 xv = *(const float4*)(xa     + (size_t)row * DD + lane * 4);
  const float4 cv = *(const float4*)(compat + (size_t)row * DD + lane * 4);
  const float4 uu = *(const float4*)(uv      + lane * 4);
  const float4 vv = *(const float4*)(uv + DD + lane * 4);

  float t1 = xv.x*uu.x + xv.y*uu.y + xv.z*uu.z + xv.w*uu.w;   // x_a . u
  float t2 = xv.x*vv.x + xv.y*vv.y + xv.z*vv.z + xv.w*vv.w;   // x_a . v
  float t3 = cv.x*uu.x + cv.y*uu.y + cv.z*uu.z + cv.w*uu.w;   // compat . u
  #pragma unroll
  for (int off = 16; off; off >>= 1) {
    t1 += __shfl_xor(t1, off, 32);
    t2 += __shfl_xor(t2, off, 32);
    t3 += __shfl_xor(t3, off, 32);
  }

  float s0 = t1 + t2, s1 = t3 + t2;                 // pre-ELU logits
  s0 = (s0 > 0.f) ? s0 : (__expf(s0) - 1.f);        // elu
  s1 = (s1 > 0.f) ? s1 : (__expf(s1) - 1.f);
  const float m  = fmaxf(s0, s1);
  const float e0 = __expf(s0 - m), e1 = __expf(s1 - m);
  const float zi = 1.f / (e0 + e1);
  const float w0 = e0 * zi, w1 = e1 * zi;

  float4 o;
  o.x = w0 * xv.x + w1 * cv.x;
  o.y = w0 * xv.y + w1 * cv.y;
  o.z = w0 * xv.z + w1 * cv.z;
  o.w = w0 * xv.w + w1 * cv.w;
  *(float4*)(out + (size_t)row * DD + lane * 4) = o;
}

// ---------------------------------------------------------------------------
extern "C" void kernel_launch(void* const* d_in, const int* in_sizes, int n_in,
                              void* d_out, int out_size, void* d_ws, size_t ws_size,
                              hipStream_t stream) {
  (void)in_sizes; (void)n_in; (void)out_size; (void)ws_size;
  const float* x_a = (const float*)d_in[0];
  const float* x_b = (const float*)d_in[1];
  const float* adj = (const float*)d_in[2];
  const float* Wq  = (const float*)d_in[3];
  const float* Wk  = (const float*)d_in[4];
  const float* Wv  = (const float*)d_in[5];
  float* out = (float*)d_out;

  // workspace layout
  float* ws_compat = (float*)d_ws;                               // B*NA*DD
  float* ws_inv    = ws_compat + (size_t)B_ * NA * DD;           // B*NA
  float* ws_uv     = ws_inv    + (size_t)B_ * NA;                // 2*DD

  const int rows = B_ * NA;                        // 16000
  const int rows_blocks = (rows * 32 + 255) / 256; // 8 rows (waves) per block

  rowsum_kernel<<<rows_blocks, 256, 0, stream>>>(adj, ws_inv);
  uv_kernel<<<1, DD, 0, stream>>>(Wq, Wk, Wv, ws_uv);
  gemm_kernel<<<dim3(NA / 16, B_), 256, 0, stream>>>(adj, x_b, ws_inv, ws_compat);
  epilogue_kernel<<<rows_blocks, 256, 0, stream>>>(x_a, ws_compat, ws_uv, out);
}
